// MultiHeadAttention_10118942949496
// MI455X (gfx1250) — matrix operand order repaired and checked by reading, not compiled, not run on hardware
//
#include <hip/hip_runtime.h>

// MHA forward for B=2, S=2048, D=1024, H=16, depth=64 on gfx1250 (wave32, WMMA).
// d_out = [out (B*S*D f32)] ++ [attn (B*H*S*S f32)]

constexpr int Bn = 2;
constexpr int Sn = 2048;
constexpr int Dn = 1024;
constexpr int Hn = 16;
constexpr int Dh = 64;     // depth per head

constexpr int ALD = Dn + 8;   // padded LDS row stride (halves): bank step 4
constexpr int SLD = Sn + 8;   // padded slog row stride (halves): bank step 4

typedef __attribute__((ext_vector_type(16))) _Float16 v16h;
typedef __attribute__((ext_vector_type(8)))  _Float16 v8h;
typedef __attribute__((ext_vector_type(4)))  _Float16 v4h;
typedef __attribute__((ext_vector_type(8)))  float    v8f;

// ---- WMMA fragment loaders (contiguous k order, half h holds k=16h..16h+15 in A and B, wave32) ----------
__device__ __forceinline__ v16h cat8(v8h lo, v8h hi) {
  return __builtin_shufflevector(lo, hi, 0, 1, 2, 3, 4, 5, 6, 7,
                                         8, 9, 10, 11, 12, 13, 14, 15);
}

// A (16x32, MxK, f16, row stride ld; ld multiple of 8): lane L<16 -> M=L,
// K=0..15; lane L>=16 -> M=L-16, K=16..31. Two b128 loads.
__device__ __forceinline__ v16h frag_a(const _Float16* base, int ld, int lane) {
  const int m  = lane & 15;
  const int kb = (lane >> 4) << 4;          // 0 or 16
  const _Float16* row = base + (size_t)m * ld + kb;
  const v8h lo = *(const v8h*)row;
  const v8h hi = *(const v8h*)(row + 8);
  return cat8(lo, hi);
}

// B (32x16, KxN) read from [N][K] storage (K contiguous, row stride ld,
// ld multiple of 8): lane L<16 -> N=L, K=0..15; lane L>=16 -> N=L-16,
// K=16..31. Two b128 loads.
__device__ __forceinline__ v16h frag_bt(const _Float16* base, int ld, int lane) {
  const int n  = lane & 15;
  const int kb = (lane >> 4) << 4;          // 0 or 16
  const _Float16* p = base + (size_t)n * ld + kb;
  const v8h lo = *(const v8h*)p;
  const v8h hi = *(const v8h*)(p + 8);
  return cat8(lo, hi);
}

__device__ __forceinline__ v8f wmma_f16(v16h a, v16h b, v8f c) {
  return __builtin_amdgcn_wmma_f32_16x16x32_f16(false, a, false, b,
                                                (short)0, c, false, false);
}

// ---- Kernel 0: W[k][n] f32 -> Wt[n][k] f16 (2 MB, stays L2-resident) -------
__global__ void __launch_bounds__(256)
transpose_w_kernel(const float* __restrict__ W, _Float16* __restrict__ Wt)
{
  __shared__ _Float16 tile[32][34];
  const int bk  = blockIdx.x * 32;          // k
  const int bnn = blockIdx.y * 32;          // n
  const int tx = threadIdx.x & 31;
  const int ty = threadIdx.x >> 5;          // 8 rows per pass
#pragma unroll
  for (int r = 0; r < 32; r += 8)
    tile[ty + r][tx] = (_Float16)W[(size_t)(bk + ty + r) * Dn + bnn + tx];
  __syncthreads();
#pragma unroll
  for (int r = 0; r < 32; r += 8)
    Wt[(size_t)(bnn + ty + r) * Dn + bk + tx] = tile[tx][ty + r];
}

// ---- Kernel 1: X @ W + bias -> head-layout f16 workspace -------------------
// mode 0: q -> Qh[b,h,s,d] * (1/sqrt(Dh))
// mode 1: k -> Kh[b,h,s,d]          (natural: logits B-frag K-run contiguous)
// mode 2: v -> VhT[b,h,d,s]         (transposed: PV  B-frag K-run contiguous)
// Entire 16x1024 A strip staged in LDS once -> barrier-free K loop.
__global__ void __launch_bounds__(256)
qkv_proj_kernel(const float* __restrict__ X, const _Float16* __restrict__ Wt,
                const float* __restrict__ bias, _Float16* __restrict__ dst,
                int mode)
{
  __shared__ _Float16 aT[16 * ALD];         // 16 x (1024+8) f16 = 33 KB
  const int tid  = threadIdx.x;
  const int wave = tid >> 5;
  const int lane = tid & 31;
  const int m0 = blockIdx.x * 16;           // row in flattened [B*S]
  const int n0 = blockIdx.y * 128;          // col in [D]
  const _Float16* wcol = Wt + (size_t)(n0 + wave * 16) * Dn;

  // stage full A strip: f32 -> f16, float4 loads, v4h stores (4096 chunks)
#pragma unroll
  for (int e = 0; e < 16; ++e) {
    const int idx = tid + e * 256;
    const int i = idx >> 8;                 // row 0..15
    const int j = (idx & 255) * 4;          // col, coalesced
    const float4 x4 = *(const float4*)(X + (size_t)(m0 + i) * Dn + j);
    v4h h4;
    h4[0] = (_Float16)x4.x; h4[1] = (_Float16)x4.y;
    h4[2] = (_Float16)x4.z; h4[3] = (_Float16)x4.w;
    *(v4h*)(aT + i * ALD + j) = h4;
  }
  __syncthreads();

  v8f acc = {};
  for (int k0 = 0; k0 < Dn; k0 += 32) {     // no barriers: free pipelining
    const v16h af = frag_a(aT + k0, ALD, lane);      // ds_load_b128 x2
    const v16h bf = frag_bt(wcol + k0, Dn, lane);    // global b128 x2 (L2)
    acc = wmma_f16(af, bf, acc);
  }

  // epilogue: bias add + scatter into head layout
  const int n_glob = n0 + wave * 16 + (lane & 15);
  const float bv = bias[n_glob];
  const int h   = n_glob / Dh;
  const int dep = n_glob % Dh;
  const int rb  = (lane >> 4) << 3;
#pragma unroll
  for (int r = 0; r < 8; ++r) {
    const int m = m0 + rb + r;
    const int bi = m / Sn;
    const int s  = m % Sn;
    float val = acc[r] + bv;
    if (mode == 0) {
      val *= 0.125f;   // fold 1/sqrt(64)
      dst[(((size_t)bi * Hn + h) * Sn + s) * Dh + dep] = (_Float16)val;
    } else if (mode == 1) {
      dst[(((size_t)bi * Hn + h) * Sn + s) * Dh + dep] = (_Float16)val;
    } else {
      dst[(((size_t)bi * Hn + h) * Dh + dep) * Sn + s] = (_Float16)val;
    }
  }
}

// ---- Kernel 2: logits -> softmax -> attn (f32 out) -> ctx = attn @ V -------
// one block per (b, h, 16-row q tile); 8 waves, padded 64 KB LDS score strip
__global__ void __launch_bounds__(256)
attention_kernel(const _Float16* __restrict__ Qh, const _Float16* __restrict__ Kh,
                 const _Float16* __restrict__ VhT, const float* __restrict__ mask,
                 float* __restrict__ attn, _Float16* __restrict__ ctx)
{
  __shared__ _Float16 slog[16 * SLD];       // 16 x (2048+8) f16 ~ 64.25 KB
  __shared__ float red[4 * 16 * 16];        // PV split-K reduction, 4 KB
  const int tid   = threadIdx.x;
  const int wave  = tid >> 5;
  const int lane  = tid & 31;
  const int nkt   = Sn / 16;                // 128 key tiles
  const int bh    = blockIdx.x / nkt;
  const int qtile = blockIdx.x % nkt;
  const int b     = bh / Hn;
  const int h     = bh % Hn;

  const _Float16* Qbase  = Qh  + ((size_t)bh * Sn + qtile * 16) * Dh;
  const _Float16* Kbase  = Kh  + (size_t)bh * Sn * Dh;
  const _Float16* VTbase = VhT + (size_t)bh * Dh * Sn;

  // Q fragments for K=64 (two 32-wide chunks); Q already scaled by 1/8
  const v16h qa0 = frag_a(Qbase,      Dh, lane);
  const v16h qa1 = frag_a(Qbase + 32, Dh, lane);

  // logits: each wave does 16 of the 128 key tiles; B-frags are b128 loads
  for (int ct = wave; ct < nkt; ct += 8) {
    const _Float16* kb = Kbase + (size_t)(ct * 16) * Dh;
    v8f acc = {};
    acc = wmma_f16(qa0, frag_bt(kb,      Dh, lane), acc);
    acc = wmma_f16(qa1, frag_bt(kb + 32, Dh, lane), acc);
    const int n  = lane & 15;
    const int rb = (lane >> 4) << 3;
#pragma unroll
    for (int r = 0; r < 8; ++r)
      slog[(rb + r) * SLD + ct * 16 + n] = (_Float16)acc[r];
  }
  __syncthreads();

  // softmax: 16 threads per row, 4-wide vectorized (v4h LDS / float4 global)
  const int row = tid >> 4;
  const int co  = (tid & 15) * 4;
  _Float16* srow = slog + row * SLD;
  const float* mrow = mask + (size_t)b * Sn;

  float rmax = -3.0e38f;
  for (int i = 0; i < Sn / 64; ++i) {
    const int col = co + (i << 6);
    const v4h h4 = *(const v4h*)(srow + col);
    const float4 mk = *(const float4*)(mrow + col);
    rmax = fmaxf(rmax, fmaxf(fmaxf((float)h4[0] + mk.x * -1e9f,
                                   (float)h4[1] + mk.y * -1e9f),
                             fmaxf((float)h4[2] + mk.z * -1e9f,
                                   (float)h4[3] + mk.w * -1e9f)));
  }
#pragma unroll
  for (int off = 1; off < 16; off <<= 1)
    rmax = fmaxf(rmax, __shfl_xor(rmax, off, 32));

  float rsum = 0.f;
  for (int i = 0; i < Sn / 64; ++i) {
    const int col = co + (i << 6);
    const v4h h4 = *(const v4h*)(srow + col);
    const float4 mk = *(const float4*)(mrow + col);
    const float e0 = __expf((float)h4[0] + mk.x * -1e9f - rmax);
    const float e1 = __expf((float)h4[1] + mk.y * -1e9f - rmax);
    const float e2 = __expf((float)h4[2] + mk.z * -1e9f - rmax);
    const float e3 = __expf((float)h4[3] + mk.w * -1e9f - rmax);
    rsum += (e0 + e1) + (e2 + e3);
    v4h eh;
    eh[0] = (_Float16)e0; eh[1] = (_Float16)e1;
    eh[2] = (_Float16)e2; eh[3] = (_Float16)e3;
    *(v4h*)(srow + col) = eh;
  }
#pragma unroll
  for (int off = 1; off < 16; off <<= 1)
    rsum += __shfl_xor(rsum, off, 32);
  const float rinv = 1.0f / rsum;

  float* arow = attn + ((size_t)bh * Sn + qtile * 16 + row) * Sn;
  for (int i = 0; i < Sn / 64; ++i) {
    const int col = co + (i << 6);
    const v4h h4 = *(const v4h*)(srow + col);
    float4 w;
    w.x = (float)h4[0] * rinv; w.y = (float)h4[1] * rinv;
    w.z = (float)h4[2] * rinv; w.w = (float)h4[3] * rinv;
    *(float4*)(arow + col) = w;                    // fp32 attn, 16B stores
    v4h wh;
    wh[0] = (_Float16)w.x; wh[1] = (_Float16)w.y;
    wh[2] = (_Float16)w.z; wh[3] = (_Float16)w.w;
    *(v4h*)(srow + col) = wh;                      // keep weights for PV GEMM
  }
  __syncthreads();

  // ctx = attn @ V : M=16, N=64, K=2048. All 8 waves: 4 N-tiles x 2 K-halves.
  {
    v8f acc = {};
    const int ncol = (wave & 3) * 16;
    const int kh0  = (wave >> 2) * (Sn / 2);
    const _Float16* vb = VTbase + (size_t)ncol * Sn + kh0;
    for (int ks = 0; ks < Sn / 2; ks += 32) {
      const v16h af = frag_a(slog + kh0 + ks, SLD, lane);  // conflict-free b128
      const v16h bf = frag_bt(vb + ks, Sn, lane);          // global b128 x2
      acc = wmma_f16(af, bf, acc);
    }
    const int n  = lane & 15;
    const int rb = (lane >> 4) << 3;
    if (wave >= 4) {
      float* r = red + (wave - 4) * 256;
#pragma unroll
      for (int rr = 0; rr < 8; ++rr) r[(rb + rr) * 16 + n] = acc[rr];
    }
    __syncthreads();
    if (wave < 4) {
      const float* r = red + wave * 256;
#pragma unroll
      for (int rr = 0; rr < 8; ++rr) {
        const float val = acc[rr] + r[(rb + rr) * 16 + n];
        const int s = qtile * 16 + rb + rr;
        ctx[((size_t)b * Sn + s) * Dn + h * Dh + ncol + n] = (_Float16)val;
      }
    }
  }
}

// ---- Kernel 3: out = ctx @ wo + bo (fp32 output) ---------------------------
__global__ void __launch_bounds__(256)
out_proj_kernel(const _Float16* __restrict__ X, const _Float16* __restrict__ Wt,
                const float* __restrict__ bias, float* __restrict__ out)
{
  __shared__ _Float16 aT[16 * ALD];         // full A strip, padded
  const int tid  = threadIdx.x;
  const int wave = tid >> 5;
  const int lane = tid & 31;
  const int m0 = blockIdx.x * 16;
  const int n0 = blockIdx.y * 128;
  const _Float16* wcol = Wt + (size_t)(n0 + wave * 16) * Dn;

  // stage full 16x1024 f16 strip: 2048 v8h chunks, 8 per thread
#pragma unroll
  for (int e = 0; e < 8; ++e) {
    const int idx = tid + e * 256;
    const int i = idx >> 7;                 // row 0..15
    const int j = (idx & 127) * 8;          // col, coalesced b128
    *(v8h*)(aT + i * ALD + j) = *(const v8h*)(X + (size_t)(m0 + i) * Dn + j);
  }
  __syncthreads();

  v8f acc = {};
  for (int k0 = 0; k0 < Dn; k0 += 32) {     // barrier-free
    const v16h af = frag_a(aT + k0, ALD, lane);
    const v16h bf = frag_bt(wcol + k0, Dn, lane);
    acc = wmma_f16(af, bf, acc);
  }

  const int n_glob = n0 + wave * 16 + (lane & 15);
  const float bv = bias[n_glob];
  const int rb = (lane >> 4) << 3;
#pragma unroll
  for (int r = 0; r < 8; ++r) {
    const int m = m0 + rb + r;
    out[(size_t)m * Dn + n_glob] = acc[r] + bv;
  }
}

// ---- Launch ----------------------------------------------------------------
extern "C" void kernel_launch(void* const* d_in, const int* in_sizes, int n_in,
                              void* d_out, int out_size, void* d_ws, size_t ws_size,
                              hipStream_t stream) {
  (void)in_sizes; (void)n_in; (void)out_size; (void)ws_size;
  const float* q    = (const float*)d_in[0];
  const float* k    = (const float*)d_in[1];
  const float* v    = (const float*)d_in[2];
  const float* mask = (const float*)d_in[3];
  const float* wq   = (const float*)d_in[4];
  const float* bq   = (const float*)d_in[5];
  const float* wk   = (const float*)d_in[6];
  const float* bk   = (const float*)d_in[7];
  const float* wv   = (const float*)d_in[8];
  const float* bv   = (const float*)d_in[9];
  const float* wo   = (const float*)d_in[10];
  const float* bo   = (const float*)d_in[11];

  float* out  = (float*)d_out;
  float* attn = out + (size_t)Bn * Sn * Dn;

  const size_t NE = (size_t)Bn * Hn * Sn * Dh;   // 4,194,304 f16 elems
  const size_t WE = (size_t)Dn * Dn;             // 1,048,576 f16 elems
  _Float16* Qh  = (_Float16*)d_ws;
  _Float16* Kh  = Qh  + NE;
  _Float16* VhT = Kh  + NE;
  _Float16* ctx = VhT + NE;
  _Float16* Wqt = ctx + NE;
  _Float16* Wkt = Wqt + WE;
  _Float16* Wvt = Wkt + WE;
  _Float16* Wot = Wvt + WE;                      // total 40 MB workspace

  const dim3 blk(256);
  const dim3 gT(Dn / 32, Dn / 32);               // 32 x 32
  transpose_w_kernel<<<gT, blk, 0, stream>>>(wq, Wqt);
  transpose_w_kernel<<<gT, blk, 0, stream>>>(wk, Wkt);
  transpose_w_kernel<<<gT, blk, 0, stream>>>(wv, Wvt);
  transpose_w_kernel<<<gT, blk, 0, stream>>>(wo, Wot);

  const dim3 gProj(Bn * Sn / 16, Dn / 128);      // 256 x 8
  qkv_proj_kernel<<<gProj, blk, 0, stream>>>(q, Wqt, bq, Qh, 0);
  qkv_proj_kernel<<<gProj, blk, 0, stream>>>(k, Wkt, bk, Kh, 1);
  qkv_proj_kernel<<<gProj, blk, 0, stream>>>(v, Wvt, bv, VhT, 2);

  const dim3 gAttn(Bn * Hn * (Sn / 16));         // 4096 blocks
  attention_kernel<<<gAttn, blk, 0, stream>>>(Qh, Kh, VhT, mask, attn, ctx);

  out_proj_kernel<<<gProj, blk, 0, stream>>>(ctx, Wot, bo, out);
}
